// MultiViewAlignmentAdapter_22436909154678
// MI455X (gfx1250) — compile-verified, matrix operand order checked by reading, not run on hardware
//
#include <hip/hip_runtime.h>
#include <math.h>

// ---------------------------------------------------------------------------
// MultiViewAlignmentAdapter recurrent gated blend, MI455X (gfx1250).
//
// Persistent kernel, one wave32 per 16x16 output tile (8 mTiles x 64 nTiles
// = 512 waves = 64 WGs x 8 waves). bf16 WMMA 16x16x32, f32 accumulate.
// All 8 waves of a WG share one nTile -> identical B fragments, so B is
// staged into LDS with double-buffered GLOBAL_LOAD_ASYNC_TO_LDS_B128
// (ASYNCcnt) and consumed via ds_load; A rides the global path per wave.
// Device-wide atomic barrier between the 5 phases of each of 512 steps.
// ---------------------------------------------------------------------------

#define HDIM 1024
#define BDIM 128
#define TDIM 512
#define CHUNK 8                 // k-tiles (of 32) staged per LDS buffer: 8 KB

typedef __bf16 bf16;
typedef __attribute__((ext_vector_type(16))) __bf16 bf16x16;
typedef __attribute__((ext_vector_type(8)))  __bf16 bf16x8;
typedef __attribute__((ext_vector_type(8)))  float  f32x8;

__device__ __forceinline__ float gelu_exact(float x) {
  return 0.5f * x * (1.0f + erff(x * 0.70710678118654752440f));
}
__device__ __forceinline__ float sigmoidf(float x) {
  return 1.0f / (1.0f + expf(-x));
}

// ---------------------------------------------------------------------------
// Weight packing: W [K x N] row-major fp32 -> bf16 WMMA B-fragments.
// Fragment (kt, nt): lane l = {idx = l&15 -> N, half = l>>4}; element e ->
// K offset (e<8 ? e : e+8) + 8*half inside the 32-wide K tile. Each lane's
// 16 bf16 are contiguous (32B) so staging/loading is pure B128 traffic.
// ---------------------------------------------------------------------------
__global__ void pack_b_kernel(const float* __restrict__ W, bf16* __restrict__ P,
                              int K, int N) {
  int tid = blockIdx.x * blockDim.x + threadIdx.x;
  int KT = K >> 5;
  int total = KT * (N >> 4) * 32;
  if (tid >= total) return;
  int lane = tid & 31;
  int frag = tid >> 5;
  int kt = frag % KT;
  int nt = frag / KT;
  int idx = lane & 15, half = lane >> 4;
  int n = nt * 16 + idx;
  bf16* dst = P + ((size_t)tid << 4);
#pragma unroll
  for (int e = 0; e < 16; ++e) {
    int k = kt * 32 + ((e < 8) ? e : e + 8) + 8 * half;
    dst[e] = (bf16)W[(size_t)k * N + n];
  }
}

__global__ void init_kernel(bf16* __restrict__ state, unsigned* __restrict__ bar) {
  int tid = blockIdx.x * blockDim.x + threadIdx.x;
  if (tid < BDIM * HDIM) state[tid] = (bf16)0.0f;
  if (tid < 8) bar[tid] = 0u;
}

// ---------------------------------------------------------------------------
// Device-wide barrier (atomic counter + epoch, s_sleep backoff).
// ---------------------------------------------------------------------------
__device__ __forceinline__ void grid_sync(unsigned* bar, unsigned nblocks) {
  __threadfence();
  __syncthreads();
  if (threadIdx.x == 0) {
    volatile unsigned* vcount = bar;
    volatile unsigned* vepoch = bar + 1;
    unsigned e = *vepoch;
    if (atomicAdd(bar, 1u) == nblocks - 1u) {
      *vcount = 0u;
      __threadfence();
      atomicAdd(bar + 1, 1u);
    } else {
      while (*vepoch == e) { __builtin_amdgcn_s_sleep(1); }
    }
  }
  __syncthreads();
  __threadfence();
}

// ---------------------------------------------------------------------------
// Async-stage one CHUNK (8 KB = CHUNK fragments) of packed B into LDS.
// 256 threads x 32B each; offset:16 covers the second B128 from same regs
// (ISA: INST_OFFSET applies to both the LDS and global address).
// ---------------------------------------------------------------------------
__device__ __forceinline__ void stage_async(const bf16* __restrict__ gsrc,
                                            bf16* ldst, int tid) {
  const bf16* g = gsrc + tid * 16;
  unsigned l = (unsigned)(uintptr_t)(ldst + tid * 16);
  asm volatile("global_load_async_to_lds_b128 %0, %1, off"
               :: "v"(l), "v"(g) : "memory");
  asm volatile("global_load_async_to_lds_b128 %0, %1, off offset:16"
               :: "v"(l), "v"(g) : "memory");
}
__device__ __forceinline__ void wait_async_all() {
  asm volatile("s_wait_asynccnt 0x0" ::: "memory");
}

// ---------------------------------------------------------------------------
// 32 K-tiles (one K=1024 chunk) of bf16 WMMA. B double-buffered through LDS
// via async loads; A per-wave from global (2 x B128 per WMMA, hoisted by
// the CHUNK-unroll). ldsB must hold 2*CHUNK*512 bf16 (16 KB).
// ---------------------------------------------------------------------------
__device__ __forceinline__ f32x8 gemm_1024(f32x8 acc,
                                           const bf16* __restrict__ A,
                                           const bf16* __restrict__ BP,
                                           int KTtotal, int ktBegin,
                                           int row, int half, int nTile,
                                           int lane, int tid, bf16* ldsB) {
  const bf16* ap = A + (size_t)row * HDIM + half * 8;
  const bf16* gb = BP + ((size_t)nTile * KTtotal + ktBegin) * 512;

  stage_async(gb, ldsB, tid);           // prologue: chunk 0 -> buffer 0
  wait_async_all();
  __syncthreads();

  int buf = 0;
#pragma unroll 1
  for (int c = 0; c < 32 / CHUNK; ++c) {
    if (c + 1 < 32 / CHUNK)             // async-prefetch next chunk
      stage_async(gb + (size_t)(c + 1) * CHUNK * 512,
                  ldsB + (buf ^ 1) * (CHUNK * 512), tid);

    const bf16* lb = ldsB + buf * (CHUNK * 512) + (lane << 4);
    __builtin_prefetch((const void*)(ap + CHUNK * 32), 0, 1);  // A stream
#pragma unroll
    for (int j = 0; j < CHUNK; ++j) {
      bf16x8 lo = *(const bf16x8*)(ap + j * 32);
      bf16x8 hi = *(const bf16x8*)(ap + j * 32 + 16);
      bf16x16 a;
#pragma unroll
      for (int i = 0; i < 8; ++i) { a[i] = lo[i]; a[i + 8] = hi[i]; }
      bf16x16 b = *(const bf16x16*)(lb + j * 512);
      acc = __builtin_amdgcn_wmma_f32_16x16x32_bf16(
          false, a, false, b, (short)0, acc, false, false);
    }
    ap += CHUNK * 32;
    wait_async_all();                   // prefetched chunk is in LDS
    __syncthreads();                    // all waves done reading old buf
    buf ^= 1;
  }
  return acc;
}

__global__ void __launch_bounds__(256, 1)
rnn_persistent(const float* __restrict__ frames,
               const float* __restrict__ ab1, const float* __restrict__ ab2,
               const float* __restrict__ gb1, const float* __restrict__ gb2,
               const bf16* __restrict__ PW1, const bf16* __restrict__ PW2,
               const bf16* __restrict__ PW3, const bf16* __restrict__ PW4,
               bf16* __restrict__ xcur, bf16* __restrict__ state,
               bf16* __restrict__ alignedB, bf16* __restrict__ h1,
               unsigned* bar, float* __restrict__ out) {
  __shared__ __align__(32) bf16 ldsB[2 * CHUNK * 512];   // 16 KB double buffer

  const int lane  = threadIdx.x & 31;
  const int wave  = threadIdx.x >> 5;
  const int idx   = lane & 15;
  const int half  = lane >> 4;
  const int mTile = wave;            // 0..7  -> batch-row tile
  const int nTile = blockIdx.x;      // 0..63 -> hidden-col tile (shared by WG)
  const int row   = mTile * 16 + idx;
  const int col   = nTile * 16 + idx;
  const int tid   = threadIdx.x;
  const int gtid  = blockIdx.x * blockDim.x + threadIdx.x;
  const int nthr  = gridDim.x * blockDim.x;

  for (int t = 0; t < TDIM; ++t) {
    // ---- P0: xcur = bf16(frames[:, t, :]) --------------------------------
    for (int i = gtid; i < BDIM * HDIM; i += nthr) {
      int b = i >> 10, h = i & (HDIM - 1);
      xcur[i] = (bf16)frames[((size_t)b * TDIM + t) * HDIM + h];
    }
    grid_sync(bar, gridDim.x);

    // ---- P1: h1 = gelu([xcur | state] @ aw1 + ab1) -----------------------
    {
      f32x8 acc = {};
      acc = gemm_1024(acc, xcur,  PW1, 64, 0,  row, half, nTile, lane, tid, ldsB);
      acc = gemm_1024(acc, state, PW1, 64, 32, row, half, nTile, lane, tid, ldsB);
      float bias = ab1[col];
#pragma unroll
      for (int r = 0; r < 8; ++r) {
        int rr = mTile * 16 + half * 8 + r;     // C/D layout: M = r + 8*half
        h1[(size_t)rr * HDIM + col] = (bf16)gelu_exact(acc[r] + bias);
      }
    }
    grid_sync(bar, gridDim.x);

    // ---- P2: aligned = h1 @ aw2 + ab2 ------------------------------------
    {
      f32x8 acc = {};
      acc = gemm_1024(acc, h1, PW2, 32, 0, row, half, nTile, lane, tid, ldsB);
      float bias = ab2[col];
#pragma unroll
      for (int r = 0; r < 8; ++r) {
        int rr = mTile * 16 + half * 8 + r;
        alignedB[(size_t)rr * HDIM + col] = (bf16)(acc[r] + bias);
      }
    }
    grid_sync(bar, gridDim.x);

    // ---- P3: h1 = gelu([xcur | aligned] @ gw1 + gb1) ---------------------
    {
      f32x8 acc = {};
      acc = gemm_1024(acc, xcur,     PW3, 64, 0,  row, half, nTile, lane, tid, ldsB);
      acc = gemm_1024(acc, alignedB, PW3, 64, 32, row, half, nTile, lane, tid, ldsB);
      float bias = gb1[col];
#pragma unroll
      for (int r = 0; r < 8; ++r) {
        int rr = mTile * 16 + half * 8 + r;
        h1[(size_t)rr * HDIM + col] = (bf16)gelu_exact(acc[r] + bias);
      }
    }
    grid_sync(bar, gridDim.x);

    // ---- P4: gate = sigmoid(h1 @ gw2 + gb2); blend; out & next state -----
    {
      f32x8 acc = {};
      acc = gemm_1024(acc, h1, PW4, 32, 0, row, half, nTile, lane, tid, ldsB);
      float bias = gb2[col];
#pragma unroll
      for (int r = 0; r < 8; ++r) {
        int rr = mTile * 16 + half * 8 + r;
        float gate = sigmoidf(acc[r] + bias);
        float al   = (float)alignedB[(size_t)rr * HDIM + col];
        float cur  = frames[((size_t)rr * TDIM + t) * HDIM + col];
        float enh  = gate * al + (1.0f - gate) * cur;
        out[((size_t)rr * TDIM + t) * HDIM + col] = enh;
        state[(size_t)rr * HDIM + col] = (bf16)enh;
      }
    }
    grid_sync(bar, gridDim.x);
  }
}

// ---------------------------------------------------------------------------
// Host launcher
// ---------------------------------------------------------------------------
extern "C" void kernel_launch(void* const* d_in, const int* in_sizes, int n_in,
                              void* d_out, int out_size, void* d_ws, size_t ws_size,
                              hipStream_t stream) {
  const float* frames = (const float*)d_in[0];
  const float* aw1 = (const float*)d_in[1];
  const float* ab1 = (const float*)d_in[2];
  const float* aw2 = (const float*)d_in[3];
  const float* ab2 = (const float*)d_in[4];
  const float* gw1 = (const float*)d_in[5];
  const float* gb1 = (const float*)d_in[6];
  const float* gw2 = (const float*)d_in[7];
  const float* gb2 = (const float*)d_in[8];
  float* out = (float*)d_out;

  char* ws = (char*)d_ws;
  size_t off = 0;
  bf16* PW1 = (bf16*)(ws + off); off += (size_t)2048 * 1024 * 2;   // 4 MB
  bf16* PW2 = (bf16*)(ws + off); off += (size_t)1024 * 1024 * 2;   // 2 MB
  bf16* PW3 = (bf16*)(ws + off); off += (size_t)2048 * 1024 * 2;   // 4 MB
  bf16* PW4 = (bf16*)(ws + off); off += (size_t)1024 * 1024 * 2;   // 2 MB
  bf16* xcur     = (bf16*)(ws + off); off += (size_t)BDIM * HDIM * 2;
  bf16* state    = (bf16*)(ws + off); off += (size_t)BDIM * HDIM * 2;
  bf16* alignedB = (bf16*)(ws + off); off += (size_t)BDIM * HDIM * 2;
  bf16* h1       = (bf16*)(ws + off); off += (size_t)BDIM * HDIM * 2;
  unsigned* bar  = (unsigned*)(ws + off); off += 256;

  pack_b_kernel<<<512, 256, 0, stream>>>(aw1, PW1, 2048, 1024);
  pack_b_kernel<<<256, 256, 0, stream>>>(aw2, PW2, 1024, 1024);
  pack_b_kernel<<<512, 256, 0, stream>>>(gw1, PW3, 2048, 1024);
  pack_b_kernel<<<256, 256, 0, stream>>>(gw2, PW4, 1024, 1024);
  init_kernel<<<512, 256, 0, stream>>>(state, bar);

  rnn_persistent<<<64, 256, 0, stream>>>(frames, ab1, ab2, gb1, gb2,
                                         PW1, PW2, PW3, PW4,
                                         xcur, state, alignedB, h1, bar, out);
}
